// Spikf_Block_s_28166395527232
// MI455X (gfx1250) — matrix operand order repaired and checked by reading, not compiled, not run on hardware
//
#include <hip/hip_runtime.h>
#include <hip/hip_bf16.h>

typedef _Float16 half_t;
typedef __attribute__((ext_vector_type(16))) _Float16 v16h;
typedef __attribute__((ext_vector_type(8)))  float    v8f;
typedef __attribute__((ext_vector_type(4)))  int      v4i;

#define TAU_INV   0.5f
#define THRESH    0.5f
#define SCALE     0.125f
#define EPS_BN    1e-5f

#if defined(__has_builtin)
#if __has_builtin(__builtin_amdgcn_global_load_async_to_lds_b128)
#define HAVE_ASYNC_B128_BUILTIN 1
#endif
#if __has_builtin(__builtin_amdgcn_s_wait_asynccnt)
#define HAVE_WAIT_ASYNC_BUILTIN 1
#endif
#endif

// ---- async global->LDS copy of 16 bytes (one lane) ----
__device__ __forceinline__ void async_copy16(half_t* ldst, const half_t* gsrc) {
#ifdef HAVE_ASYNC_B128_BUILTIN
  __builtin_amdgcn_global_load_async_to_lds_b128(
      (__attribute__((address_space(1))) v4i*)(unsigned long long)(size_t)gsrc,
      (__attribute__((address_space(3))) v4i*)(unsigned)(size_t)ldst, 0, 0);
#else
  unsigned l = (unsigned)(size_t)ldst;
  asm volatile("global_load_async_to_lds_b128 %0, %1, off" :: "v"(l), "v"(gsrc) : "memory");
#endif
}

__device__ __forceinline__ void async_wait0() {
#ifdef HAVE_WAIT_ASYNC_BUILTIN
  __builtin_amdgcn_s_wait_asynccnt(0);
#else
  asm volatile("s_wait_asynccnt 0x0" ::: "memory");
#endif
}

// ---------------- WMMA fragment loaders (16x16x32 f16, wave32) ----------------
// A: 16x32 (MxK), row-major source with leading dim lda.
__device__ __forceinline__ v16h load_A_frag(const half_t* __restrict__ A, int lda, int lane) {
  int m = lane & 15, hh = lane >> 4;
  const half_t* row = A + (size_t)m * lda;
  v16h a;
#pragma unroll
  for (int j = 0; j < 4; ++j) {
    int k0 = hh * 16 + 2 * j;
    a[2 * j]     = row[k0];
    a[2 * j + 1] = row[k0 + 1];
    int k1 = hh * 16 + 8 + 2 * j;
    a[8 + 2 * j]     = row[k1];
    a[8 + 2 * j + 1] = row[k1 + 1];
  }
  return a;
}

__device__ __forceinline__ v16h load_A_frag_guard(const half_t* __restrict__ A, int lda, int rows, int lane) {
  int m = lane & 15, hh = lane >> 4;
  v16h a;
  if (m < rows) {
    const half_t* row = A + (size_t)m * lda;
#pragma unroll
    for (int j = 0; j < 4; ++j) {
      int k0 = hh * 16 + 2 * j;
      a[2 * j]     = row[k0];
      a[2 * j + 1] = row[k0 + 1];
      int k1 = hh * 16 + 8 + 2 * j;
      a[8 + 2 * j]     = row[k1];
      a[8 + 2 * j + 1] = row[k1 + 1];
    }
  } else {
#pragma unroll
    for (int j = 0; j < 16; ++j) a[j] = (half_t)0.f;
  }
  return a;
}

// B: 32x16 (KxN), source "weight-style" row-major [N x K] (B[k][n] = W[n*ldw + k]).
__device__ __forceinline__ v16h load_B_fragW(const half_t* __restrict__ W, int ldw, int lane) {
  int n = lane & 15, hh = lane >> 4;
  const half_t* col = W + (size_t)n * ldw;
  v16h b;
#pragma unroll
  for (int j = 0; j < 8; ++j) {
    int k = hh * 16 + 2 * j;
    b[2 * j]     = col[k];
    b[2 * j + 1] = col[k + 1];
  }
  return b;
}

// B: 32x16 (KxN), source row-major [K x N] with leading dim ldb.
__device__ __forceinline__ v16h load_B_rowKN(const half_t* __restrict__ Bm, int ldb, int lane) {
  int n = lane & 15, hh = lane >> 4;
  v16h b;
#pragma unroll
  for (int j = 0; j < 8; ++j) {
    int k = hh * 16 + 2 * j;
    b[2 * j]     = Bm[(size_t)k * ldb + n];
    b[2 * j + 1] = Bm[(size_t)(k + 1) * ldb + n];
  }
  return b;
}

// ---------------- elementwise cast ----------------
__global__ void k_cast_f32_to_f16(const float* __restrict__ src, half_t* __restrict__ dst, size_t n) {
  size_t i = (size_t)blockIdx.x * blockDim.x + threadIdx.x;
  if (i < n) dst[i] = (half_t)src[i];
}

// ---------------- GEMM: Y[M,O] = A[M,K](f16) @ W[O,K]^T(f16) + bias, fp32 out ----------------
// grid.x = M/16, grid.y = O_tiles/8, blockDim = 256 (8 waves, one 16x16 tile each).
// The 16-row A band (contiguous 16*K halfs) is async-staged to LDS once per block.
#define KMAX 1536
__global__ void k_gemm_bias(const half_t* __restrict__ A, const half_t* __restrict__ W,
                            const float* __restrict__ bias, float* __restrict__ Y,
                            int M, int K, int O) {
  __shared__ half_t As[16 * KMAX];
  const int lane = threadIdx.x & 31;
  const int wave = threadIdx.x >> 5;
  const int m0 = blockIdx.x * 16;
  const int o0 = (blockIdx.y * 8 + wave) * 16;

  const half_t* Ag = A + (size_t)m0 * K;
  for (int i = threadIdx.x; i < 2 * K; i += 256)       // 16-byte chunks
    async_copy16(&As[i * 8], &Ag[i * 8]);
  async_wait0();
  __syncthreads();

  const half_t* Wrow = W + (size_t)o0 * K;
  v8f acc = {};
  for (int k = 0; k < K; k += 32) {
    __builtin_prefetch(Wrow + k + 128, 0, 0);
    v16h a = load_A_frag(As + k, K, lane);
    v16h b = load_B_fragW(Wrow + k, K, lane);
    acc = __builtin_amdgcn_wmma_f32_16x16x32_f16(false, a, false, b, (short)0, acc, false, false);
  }
  int n = lane & 15, hh = lane >> 4;
  float bs = bias[o0 + n];
#pragma unroll
  for (int j = 0; j < 8; ++j) {
    int m = m0 + hh * 8 + j;
    Y[(size_t)m * O + o0 + n] = acc[j] + bs;
  }
}

// ---------------- BN statistics (deterministic): one block per channel ----------------
__global__ void k_bn_stats(const float* __restrict__ Y, int M, int C,
                           const float* __restrict__ gamma, const float* __restrict__ beta,
                           float* __restrict__ sa, float* __restrict__ sb) {
  int c = blockIdx.x;
  float s = 0.f, s2 = 0.f;
  for (int m = threadIdx.x; m < M; m += blockDim.x) {
    float v = Y[(size_t)m * C + c];
    s += v; s2 += v * v;
  }
  __shared__ float red[512];
  red[threadIdx.x] = s;
  red[256 + threadIdx.x] = s2;
  __syncthreads();
  for (int w = 128; w > 0; w >>= 1) {
    if ((int)threadIdx.x < w) {
      red[threadIdx.x] += red[threadIdx.x + w];
      red[256 + threadIdx.x] += red[256 + threadIdx.x + w];
    }
    __syncthreads();
  }
  if (threadIdx.x == 0) {
    float invM = 1.f / (float)M;
    float mean = red[0] * invM;
    float var  = red[256] * invM - mean * mean;
    float g    = gamma[c] * rsqrtf(var + EPS_BN);
    sa[c] = g;
    sb[c] = beta[c] - mean * g;
  }
}

// ---------------- BN + temporal LIF -> f16 spikes ----------------
__global__ void k_bn_lif(const float* __restrict__ Y,
                         const float* __restrict__ sa, const float* __restrict__ sb,
                         half_t* __restrict__ S, size_t P, int C) {
  size_t i = (size_t)blockIdx.x * blockDim.x + threadIdx.x;
  if (i >= P) return;
  int c = (int)(i % (size_t)C);
  float a = sa ? sa[c] : 1.f;
  float b = sb ? sb[c] : 0.f;
  float mem = 0.f;
#pragma unroll
  for (int t = 0; t < 4; ++t) {
    float xt = a * Y[(size_t)t * P + i] + b;
    mem = mem + (xt - mem) * TAU_INV;
    float s = (mem - THRESH) >= 0.f ? 1.f : 0.f;
    mem = mem * (1.f - s);
    S[(size_t)t * P + i] = (half_t)s;
  }
}

// ---------------- BN + temporal LIF + residual add ----------------
__global__ void k_bn_lif_res(const float* __restrict__ Y,
                             const float* __restrict__ sa, const float* __restrict__ sb,
                             const float* __restrict__ xin,
                             float* __restrict__ xout, half_t* __restrict__ xout_h,
                             size_t P, int C) {
  size_t i = (size_t)blockIdx.x * blockDim.x + threadIdx.x;
  if (i >= P) return;
  int c = (int)(i % (size_t)C);
  float a = sa[c], b = sb[c];
  float mem = 0.f;
#pragma unroll
  for (int t = 0; t < 4; ++t) {
    float xt = a * Y[(size_t)t * P + i] + b;
    mem = mem + (xt - mem) * TAU_INV;
    float s = (mem - THRESH) >= 0.f ? 1.f : 0.f;
    mem = mem * (1.f - s);
    float xo = xin[(size_t)t * P + i] + s;
    xout[(size_t)t * P + i] = xo;
    if (xout_h) xout_h[(size_t)t * P + i] = (half_t)xo;
  }
}

// ---------------- Attention: r = (q k^T * s) v * s, per (t,b,h) ----------------
#define NSEQ   196
#define NPAD   208   // 13 * 16
#define KPAD   224   // 7 * 32
#define NTILES 13

// grid.x = T*B*H, grid.y = 4 (row-tile groups), blockDim = 128 (4 waves)
__global__ void k_attn(const half_t* __restrict__ Sq, const half_t* __restrict__ Sk,
                       const half_t* __restrict__ Sv, float* __restrict__ R,
                       int B, int C, int Hh) {
  const int d = C / Hh;                 // 32
  int idx = blockIdx.x;
  int h  = idx % Hh;
  int tb = idx / Hh;                    // t*B + b
  const int tid = threadIdx.x, lane = tid & 31, wave = tid >> 5;

  __shared__ half_t ks[NPAD * 32];
  __shared__ half_t vs[KPAD * 32];
  __shared__ half_t sbuf[4 * 16 * KPAD];

  const size_t base = (size_t)tb * NSEQ * C + (size_t)h * d;

  // async-stage K and V rows (16-byte chunks: 4 per 32-half row)
  for (int i = tid; i < NSEQ * 4; i += blockDim.x) {
    int r = i >> 2, c8 = (i & 3) * 8;
    async_copy16(&ks[r * 32 + c8], &Sk[base + (size_t)r * C + c8]);
    async_copy16(&vs[r * 32 + c8], &Sv[base + (size_t)r * C + c8]);
  }
  // zero the padding rows with plain DS stores
  for (int i = tid; i < (NPAD - NSEQ) * 32; i += blockDim.x)
    ks[NSEQ * 32 + i] = (half_t)0.f;
  for (int i = tid; i < (KPAD - NSEQ) * 32; i += blockDim.x)
    vs[NSEQ * 32 + i] = (half_t)0.f;
  async_wait0();
  __syncthreads();

  const int it = blockIdx.y * 4 + wave;  // row tile index
  if (it >= NTILES) return;

  half_t* sw = sbuf + wave * 16 * KPAD;
  const int n = lane & 15, hh = lane >> 4;

  // q fragment straight from global (rows guarded)
  v16h aq = load_A_frag_guard(Sq + base + (size_t)(it * 16) * C, C, NSEQ - it * 16, lane);

  // ---- S row-band = q kT * SCALE, stored f16 in per-wave LDS ----
  for (int jt = 0; jt < NTILES; ++jt) {
    v16h bk = load_B_fragW(ks + jt * 16 * 32, 32, lane);
    v8f cs = {};
    cs = __builtin_amdgcn_wmma_f32_16x16x32_f16(false, aq, false, bk, (short)0, cs, false, false);
#pragma unroll
    for (int j = 0; j < 8; ++j)
      sw[(hh * 8 + j) * KPAD + jt * 16 + n] = (half_t)(cs[j] * SCALE);
  }
  // zero the K-pad columns 208..223
#pragma unroll
  for (int j = 0; j < 8; ++j)
    sw[(hh * 8 + j) * KPAD + NPAD + n] = (half_t)0.f;

  // ---- r row-band = S v * SCALE ----
  for (int n2 = 0; n2 < 2; ++n2) {
    v8f acc = {};
#pragma unroll
    for (int kc = 0; kc < KPAD / 32; ++kc) {
      v16h as = load_A_frag(sw + kc * 32, KPAD, lane);
      v16h bv = load_B_rowKN(vs + kc * 32 * 32 + n2 * 16, 32, lane);
      acc = __builtin_amdgcn_wmma_f32_16x16x32_f16(false, as, false, bv, (short)0, acc, false, false);
    }
#pragma unroll
    for (int j = 0; j < 8; ++j) {
      int row = it * 16 + hh * 8 + j;
      if (row < NSEQ)
        R[base + (size_t)row * C + n2 * 16 + n] = acc[j] * SCALE;
    }
  }
}

// ================================================================
extern "C" void kernel_launch(void* const* d_in, const int* in_sizes, int n_in,
                              void* d_out, int out_size, void* d_ws, size_t ws_size,
                              hipStream_t stream) {
  (void)in_sizes; (void)n_in; (void)out_size; (void)ws_size;
  const int T = 4, B = 32, N = 196, C = 384, Hd = 1536, Hh = 12;
  const int M = T * B * N;                       // 25088 rows, /16 = 1568
  const size_t ELEM_X = (size_t)M * C;           // 9,633,792
  const size_t ELEM_H = (size_t)M * Hd;          // 38,535,168
  const size_t P  = (size_t)B * N * C;
  const size_t PH = (size_t)B * N * Hd;

  const float* x  = (const float*)d_in[0];
  const float* Wq = (const float*)d_in[1],  *bq = (const float*)d_in[2];
  const float* gq = (const float*)d_in[3],  *eq = (const float*)d_in[4];
  const float* Wk = (const float*)d_in[5],  *bk = (const float*)d_in[6];
  const float* gk = (const float*)d_in[7],  *ek = (const float*)d_in[8];
  const float* Wv = (const float*)d_in[9],  *bv = (const float*)d_in[10];
  const float* gv = (const float*)d_in[11], *ev = (const float*)d_in[12];
  const float* Wp = (const float*)d_in[13], *bp = (const float*)d_in[14];
  const float* gp = (const float*)d_in[15], *ep = (const float*)d_in[16];
  const float* W1 = (const float*)d_in[17], *b1 = (const float*)d_in[18];
  const float* g1 = (const float*)d_in[19], *e1 = (const float*)d_in[20];
  const float* W2 = (const float*)d_in[21], *b2 = (const float*)d_in[22];
  const float* g2 = (const float*)d_in[23], *e2 = (const float*)d_in[24];
  float* out = (float*)d_out;

  char* ws = (char*)d_ws;
  size_t off = 0;
  auto alloc = [&](size_t bytes) -> char* {
    char* p = ws + off;
    off = (off + bytes + 255) & ~(size_t)255;
    return p;
  };
  half_t* XH  = (half_t*)alloc(ELEM_X * 2);            // x f16; reused as x2 f16
  half_t* WQH = (half_t*)alloc((size_t)C * C * 2);
  half_t* WKH = (half_t*)alloc((size_t)C * C * 2);
  half_t* WVH = (half_t*)alloc((size_t)C * C * 2);
  half_t* WPH = (half_t*)alloc((size_t)C * C * 2);
  half_t* W1H = (half_t*)alloc((size_t)Hd * C * 2);
  half_t* W2H = (half_t*)alloc((size_t)C * Hd * 2);
  float*  YB  = (float*)alloc(ELEM_H * 4);             // GEMM scratch (all stages) + attn out
  half_t* SBK = (half_t*)alloc(ELEM_H * 2);            // Sq|Sk|Sv|Rs, reused as S1
  float*  X2  = (float*)alloc(ELEM_X * 4);
  float*  SA  = (float*)alloc((size_t)Hd * 4);
  float*  SB  = (float*)alloc((size_t)Hd * 4);

  half_t* Sq = SBK;
  half_t* Sk = SBK + ELEM_X;
  half_t* Sv = SBK + 2 * ELEM_X;
  half_t* Rs = SBK + 3 * ELEM_X;
  half_t* S1 = SBK;

  auto cdiv = [](size_t a, size_t b) { return (unsigned)((a + b - 1) / b); };

  // ---- casts ----
  k_cast_f32_to_f16<<<cdiv(ELEM_X, 256), 256, 0, stream>>>(x, XH, ELEM_X);
  k_cast_f32_to_f16<<<cdiv((size_t)C * C, 256), 256, 0, stream>>>(Wq, WQH, (size_t)C * C);
  k_cast_f32_to_f16<<<cdiv((size_t)C * C, 256), 256, 0, stream>>>(Wk, WKH, (size_t)C * C);
  k_cast_f32_to_f16<<<cdiv((size_t)C * C, 256), 256, 0, stream>>>(Wv, WVH, (size_t)C * C);
  k_cast_f32_to_f16<<<cdiv((size_t)C * C, 256), 256, 0, stream>>>(Wp, WPH, (size_t)C * C);
  k_cast_f32_to_f16<<<cdiv((size_t)Hd * C, 256), 256, 0, stream>>>(W1, W1H, (size_t)Hd * C);
  k_cast_f32_to_f16<<<cdiv((size_t)C * Hd, 256), 256, 0, stream>>>(W2, W2H, (size_t)C * Hd);

  dim3 gC(M / 16, C / 16 / 8);     // (1568, 3)
  dim3 gH(M / 16, Hd / 16 / 8);    // (1568, 12)

  // ---- q / k / v branches: linear -> bn -> lif ----
  struct { const half_t* Wh; const float* b; const float* g; const float* e; half_t* S; }
  qkv[3] = { {WQH, bq, gq, eq, Sq}, {WKH, bk, gk, ek, Sk}, {WVH, bv, gv, ev, Sv} };
  for (int l = 0; l < 3; ++l) {
    k_gemm_bias<<<gC, 256, 0, stream>>>(XH, qkv[l].Wh, qkv[l].b, YB, M, C, C);
    k_bn_stats<<<C, 256, 0, stream>>>(YB, M, C, qkv[l].g, qkv[l].e, SA, SB);
    k_bn_lif<<<cdiv(P, 256), 256, 0, stream>>>(YB, SA, SB, qkv[l].S, P, C);
  }

  // ---- SSA attention (writes R into YB) ----
  k_attn<<<dim3(T * B * Hh, 4), 128, 0, stream>>>(Sq, Sk, Sv, YB, B, C, Hh);

  // ---- attn_lif (no BN) ----
  k_bn_lif<<<cdiv(P, 256), 256, 0, stream>>>(YB, nullptr, nullptr, Rs, P, C);

  // ---- proj: linear -> bn -> lif ; residual-1: x2 = x + spike ----
  k_gemm_bias<<<gC, 256, 0, stream>>>(Rs, WPH, bp, YB, M, C, C);
  k_bn_stats<<<C, 256, 0, stream>>>(YB, M, C, gp, ep, SA, SB);
  k_bn_lif_res<<<cdiv(P, 256), 256, 0, stream>>>(YB, SA, SB, x, X2, XH, P, C);

  // ---- fc1: linear -> bn -> lif ----
  k_gemm_bias<<<gH, 256, 0, stream>>>(XH, W1H, b1, YB, M, C, Hd);
  k_bn_stats<<<Hd, 256, 0, stream>>>(YB, M, Hd, g1, e1, SA, SB);
  k_bn_lif<<<cdiv(PH, 256), 256, 0, stream>>>(YB, SA, SB, S1, PH, Hd);

  // ---- fc2: linear -> bn -> lif ; residual-2 -> out ----
  k_gemm_bias<<<gC, 256, 0, stream>>>(S1, W2H, b2, YB, M, Hd, C);
  k_bn_stats<<<C, 256, 0, stream>>>(YB, M, C, g2, e2, SA, SB);
  k_bn_lif_res<<<cdiv(P, 256), 256, 0, stream>>>(YB, SA, SB, X2, out, nullptr, P, C);
}
